// MoCo_27719718928955
// MI455X (gfx1250) — matrix operand order repaired and checked by reading, not compiled, not run on hardware
//
#include <hip/hip_runtime.h>
#include <hip/hip_bf16.h>

typedef __attribute__((ext_vector_type(16))) _Float16 v16h;
typedef __attribute__((ext_vector_type(8)))  _Float16 v8h;
typedef __attribute__((ext_vector_type(8)))  float    v8f;

#define NROW  1024
#define DDIM  128
#define KCOLS 65536
#define ALPHA_C 1.0f
#define BETA_C  1.0f

// ---------------- prep kernels ----------------

// q (f32, [2,1024,128]) -> f16 copy, same layout
__global__ void moco_conv_q_f16(const float* __restrict__ q, _Float16* __restrict__ qh) {
    size_t i = (size_t)blockIdx.x * 256 + threadIdx.x;   // grid covers 2*1024*128
    qh[i] = (_Float16)q[i];
}

// queue (f32, [128,65536]) -> queueT f16 [65536,128]; LDS-tiled 32x32 transpose
__global__ void moco_transpose_queue_f16(const float* __restrict__ queue,
                                         _Float16* __restrict__ qt) {
    __shared__ float tile[32][33];
    const int cBase = blockIdx.x * 32;      // column (K) base
    const int dBase = blockIdx.y * 32;      // row (D) base
    const int tx = threadIdx.x & 31;
    const int ty = threadIdx.x >> 5;        // 0..7
    #pragma unroll
    for (int r = ty; r < 32; r += 8)
        tile[r][tx] = queue[(size_t)(dBase + r) * KCOLS + (cBase + tx)];
    __syncthreads();
    #pragma unroll
    for (int r = ty; r < 32; r += 8)
        qt[(size_t)(cBase + r) * DDIM + (dBase + tx)] = (_Float16)tile[tx][r];
}

__global__ void moco_zero(float* __restrict__ p, int n) {
    int i = blockIdx.x * 256 + threadIdx.x;
    if (i < n) p[i] = 0.0f;
}

// ---------------- WMMA fragment helpers (ISA 7.12.2 wave32 layouts) ----------------

// A fragment 16x32 f16: lane m=lane&15, hi=lane>>4.
// a[0..7]  <-> VGPR0..3 <-> K = kk + 8*hi  + {0..7}
// a[8..15] <-> VGPR4..7 <-> K = kk + 16 + 8*hi + {0..7}
__device__ inline v16h load_a_frag(const _Float16* __restrict__ rowp, int kk, int hi) {
    const _Float16* p = rowp + kk + 8 * hi;
    v8h lo  = *(const v8h*)(p);
    v8h hi8 = *(const v8h*)(p + 16);
    return __builtin_shufflevector(lo, hi8, 0,1,2,3,4,5,6,7,8,9,10,11,12,13,14,15);
}

// B fragment 32x16 f16: lane n=lane&15, hi=lane>>4.
// b[0..7] <-> K = kk + 8*hi + {0..7}; b[8..15] <-> K = kk + 16 + 8*hi + {0..7}
__device__ inline v16h load_b_frag(const _Float16* __restrict__ colrowp, int kk, int hi) {
    const _Float16* p = colrowp + kk + 8 * hi;
    v8h b0 = *(const v8h*)(p);
    v8h b1 = *(const v8h*)(p + 16);
    return __builtin_shufflevector(b0, b1, 0,1,2,3,4,5,6,7,8,9,10,11,12,13,14,15);
}

__device__ inline void zero_v8f(v8f& v) {
    #pragma unroll
    for (int t = 0; t < 8; ++t) v[t] = 0.0f;
}

// ---------------- qdq: per-row sums over q0 @ queue ----------------
// grid (16, 64), block 256 (8 waves). Wave owns row-tile blockIdx.y, streams 32 col-tiles.
// Software-pipelined: next tile's 4 B fragments (8x global_load_b128 in a clause) are
// in flight while the current tile's 4 WMMAs + exp tail execute.
__global__ void moco_qdq_kernel(const _Float16* __restrict__ qh,     // view0 rows used
                                const _Float16* __restrict__ qth,    // [KCOLS][DDIM]
                                float* __restrict__ accQA,
                                float* __restrict__ accQB) {
    const int lane = threadIdx.x & 31;
    const int wave = threadIdx.x >> 5;
    const int hi   = lane >> 4;
    const int l15  = lane & 15;
    const int rowTile = blockIdx.y;
    const int m = rowTile * 16 + l15;

    const _Float16* arow = qh + (size_t)m * DDIM;   // q[0] rows only (memoized qdq)
    v16h afrag[4];
    #pragma unroll
    for (int s = 0; s < 4; ++s) afrag[s] = load_a_frag(arow, s * 32, hi);

    v8f accA, accB;
    zero_v8f(accA); zero_v8f(accB);

    const int ct0 = blockIdx.x * 256 + wave;        // this wave's first col-tile
    v16h bcur[4];
    {
        const _Float16* brow = qth + (size_t)(ct0 * 16 + l15) * DDIM;
        #pragma unroll
        for (int s = 0; s < 4; ++s) bcur[s] = load_b_frag(brow, s * 32, hi);
    }

    for (int tt = 0; tt < 32; ++tt) {
        // prefetch / preload next tile's B fragments (wrap on last iter, result unused)
        const int ctn = ct0 + ((tt + 1) & 31) * 8;
        const _Float16* brown = qth + (size_t)(ctn * 16 + l15) * DDIM;
        __builtin_prefetch(qth + (size_t)((ct0 + ((tt + 2) & 31) * 8) * 16 + l15) * DDIM, 0, 1);
        v16h bnext[4];
        #pragma unroll
        for (int s = 0; s < 4; ++s) bnext[s] = load_b_frag(brown, s * 32, hi);

        v8f c; zero_v8f(c);
        #pragma unroll
        for (int s = 0; s < 4; ++s)
            c = __builtin_amdgcn_wmma_f32_16x16x32_f16(false, afrag[s], false, bcur[s],
                                                       (short)0, c, false, false);
        #pragma unroll
        for (int j = 0; j < 8; ++j) {
            float z = 2.0f * c[j] - 2.0f;      // z = -(2 - 2*dot), z in [-4, ~0]
            float e = __expf(z);
            accA[j] += e;
            accB[j] += z * e;
        }
        #pragma unroll
        for (int s = 0; s < 4; ++s) bcur[s] = bnext[s];
    }

    // reduce across 16 lanes inside each half (xor masks < 16 keep halves separate)
    #pragma unroll
    for (int j = 0; j < 8; ++j) {
        float a = accA[j], b = accB[j];
        for (int mk = 1; mk < 16; mk <<= 1) {
            a += __shfl_xor(a, mk, 32);
            b += __shfl_xor(b, mk, 32);
        }
        if (l15 == 0) {
            int row = rowTile * 16 + j + 8 * hi;
            atomicAdd(&accQA[row], a);
            atomicAdd(&accQB[row], b);
        }
    }
}

// ---------------- intra: per-row sums over qx @ qx^T, diagonal excluded ----------------
// grid (64, 2), block 256 (8 waves). Wave streams 8 of the 64 col-tiles, same pipelining.
__global__ void moco_intra_kernel(const _Float16* __restrict__ qh,
                                  float* __restrict__ accIA,
                                  float* __restrict__ accIB) {
    const int lane = threadIdx.x & 31;
    const int wave = threadIdx.x >> 5;
    const int hi   = lane >> 4;
    const int l15  = lane & 15;
    const int rowTile = blockIdx.x;
    const int view    = blockIdx.y;
    const _Float16* qv = qh + (size_t)view * NROW * DDIM;
    const int m = rowTile * 16 + l15;

    const _Float16* arow = qv + (size_t)m * DDIM;
    v16h afrag[4];
    #pragma unroll
    for (int s = 0; s < 4; ++s) afrag[s] = load_a_frag(arow, s * 32, hi);

    v8f accA, accB;
    zero_v8f(accA); zero_v8f(accB);

    v16h bcur[4];
    {
        const _Float16* brow = qv + (size_t)(wave * 16 + l15) * DDIM;
        #pragma unroll
        for (int s = 0; s < 4; ++s) bcur[s] = load_b_frag(brow, s * 32, hi);
    }

    for (int tt = 0; tt < 8; ++tt) {
        const int ct  = wave + tt * 8;                     // 0..63
        const int ctn = wave + ((tt + 1) & 7) * 8;
        const _Float16* brown = qv + (size_t)(ctn * 16 + l15) * DDIM;
        v16h bnext[4];
        #pragma unroll
        for (int s = 0; s < 4; ++s) bnext[s] = load_b_frag(brown, s * 32, hi);

        v8f c; zero_v8f(c);
        #pragma unroll
        for (int s = 0; s < 4; ++s)
            c = __builtin_amdgcn_wmma_f32_16x16x32_f16(false, afrag[s], false, bcur[s],
                                                       (short)0, c, false, false);
        #pragma unroll
        for (int j = 0; j < 8; ++j) {
            int gr = rowTile * 16 + j + 8 * hi;
            int gc = ct * 16 + l15;
            if (gr != gc) {                      // off_diag semantics
                float z = 2.0f * c[j] - 2.0f;
                float e = __expf(z);
                accA[j] += e;
                accB[j] += z * e;
            }
        }
        #pragma unroll
        for (int s = 0; s < 4; ++s) bcur[s] = bnext[s];
    }

    #pragma unroll
    for (int j = 0; j < 8; ++j) {
        float a = accA[j], b = accB[j];
        for (int mk = 1; mk < 16; mk <<= 1) {
            a += __shfl_xor(a, mk, 32);
            b += __shfl_xor(b, mk, 32);
        }
        if (l15 == 0) {
            int row = view * NROW + rowTile * 16 + j + 8 * hi;
            atomicAdd(&accIA[row], a);
            atomicAdd(&accIB[row], b);
        }
    }
}

// ---------------- loss1: mean ||q[x,i] - k[1-x,i]||^2 (f32 exact) ----------------
__global__ void moco_loss1_kernel(const float* __restrict__ q,
                                  const float* __restrict__ k,
                                  float* __restrict__ accL1) {
    __shared__ float red[256];
    int idx = blockIdx.x * 256 + threadIdx.x;    // 0..2047
    int x = idx >> 10, i = idx & 1023;
    const float* qp = q + ((size_t)x * NROW + i) * DDIM;
    const float* kp = k + ((size_t)(1 - x) * NROW + i) * DDIM;
    float v = 0.0f;
    #pragma unroll 4
    for (int d = 0; d < DDIM; ++d) {
        float t = qp[d] - kp[d];
        v += t * t;
    }
    red[threadIdx.x] = v;
    __syncthreads();
    for (int s = 128; s > 0; s >>= 1) {
        if (threadIdx.x < s) red[threadIdx.x] += red[threadIdx.x + s];
        __syncthreads();
    }
    if (threadIdx.x == 0) atomicAdd(accL1, red[0]);
}

// ---------------- finalize: combine per-row A/B, write 2 scalars ----------------
__global__ void moco_finalize_kernel(const float* __restrict__ accQA,
                                     const float* __restrict__ accQB,
                                     const float* __restrict__ accIA,
                                     const float* __restrict__ accIB,
                                     const float* __restrict__ accL1,
                                     float* __restrict__ out) {
    __shared__ float red[256];
    float sum = 0.0f;
    for (int idx = threadIdx.x; idx < 2 * NROW; idx += 256) {
        int x = idx >> 10, i = idx & 1023;
        float A = accQA[i] + accIA[x * NROW + i];   // queue partials shared by both views
        float B = accQB[i] + accIB[x * NROW + i];
        sum += B / A;                               // = sum_j z_j softmax(z)_j
    }
    red[threadIdx.x] = sum;
    __syncthreads();
    for (int s = 128; s > 0; s >>= 1) {
        if (threadIdx.x < s) red[threadIdx.x] += red[threadIdx.x + s];
        __syncthreads();
    }
    if (threadIdx.x == 0) {
        out[0] = ALPHA_C * accL1[0] / (float)(2 * NROW);   // loss1 / V
        out[1] = BETA_C  * red[0]   / (float)(2 * NROW);   // loss2 / V
    }
}

// ---------------- launch ----------------
extern "C" void kernel_launch(void* const* d_in, const int* in_sizes, int n_in,
                              void* d_out, int out_size, void* d_ws, size_t ws_size,
                              hipStream_t stream) {
    const float* q     = (const float*)d_in[0];   // [2,1024,128]
    const float* k     = (const float*)d_in[1];   // [2,1024,128]
    const float* queue = (const float*)d_in[2];   // [128,65536]

    // workspace layout
    _Float16* qh  = (_Float16*)d_ws;                                   // 512 KB
    _Float16* qth = (_Float16*)((char*)d_ws + (size_t)512 * 1024);     // 16 MB
    float* acc    = (float*)((char*)d_ws + (size_t)512 * 1024 + (size_t)16 * 1024 * 1024);
    float* accQA = acc;            // 1024
    float* accQB = acc + 1024;     // 1024
    float* accIA = acc + 2048;     // 2048
    float* accIB = acc + 4096;     // 2048
    float* accL1 = acc + 6144;     // 1
    const int nAcc = 6145;

    moco_conv_q_f16<<<(2 * NROW * DDIM) / 256, 256, 0, stream>>>(q, qh);
    moco_transpose_queue_f16<<<dim3(KCOLS / 32, DDIM / 32), 256, 0, stream>>>(queue, qth);
    moco_zero<<<(nAcc + 255) / 256, 256, 0, stream>>>(acc, nAcc);

    moco_qdq_kernel<<<dim3(16, 64), 256, 0, stream>>>(qh, qth, accQA, accQB);
    moco_intra_kernel<<<dim3(64, 2), 256, 0, stream>>>(qh, accIA, accIB);
    moco_loss1_kernel<<<8, 256, 0, stream>>>(q, k, accL1);

    moco_finalize_kernel<<<1, 256, 0, stream>>>(accQA, accQB, accIA, accIB, accL1,
                                                (float*)d_out);
}
